// projector_12936441495640
// MI455X (gfx1250) — matrix operand order repaired and checked by reading, not compiled, not run on hardware
//
#include <hip/hip_runtime.h>
#include <hip/hip_bf16.h>

// ---------------------------------------------------------------------------
// Problem constants (match reference)
// ---------------------------------------------------------------------------
#define NB   8      // batch
#define NC   512    // channels
#define NL   4096   // sequence length
#define NHW  64     // spatial side (64*64 = 4096)
#define NH   8      // heads
#define ND   64     // head dim
#define PQ   4096   // q positions (stride 1)
#define PK   1024   // k/v positions (stride 2 -> 32*32)

typedef __attribute__((ext_vector_type(16))) _Float16 v16h;
typedef __attribute__((ext_vector_type(8)))  _Float16 v8h;
typedef __attribute__((ext_vector_type(8)))  float    v8f;

#define WMMA_F16(A, B, C) \
  __builtin_amdgcn_wmma_f32_16x16x32_f16(false, (A), false, (B), (short)0, (C), false, false)

// ---------------------------------------------------------------------------
// WMMA fragment loaders (wave32, 16x16x32 f16)
//
// A-matrix 16x32 f16, per ISA table: lanes 0-15 hold row M=lane with
// K = {0..7, 16..23}; lanes 16-31 hold row M=lane-16 with K = {8..15, 24..31}.
// Source is row-major [16 x K] with row stride `stride` (halves).
// ---------------------------------------------------------------------------
__device__ __forceinline__ v16h load_fragA(const _Float16* __restrict__ base, int stride) {
  int lane = threadIdx.x & 31;
  int r    = lane & 15;
  int hi   = (lane >> 4) << 3;            // 0 or 8
  const _Float16* p0 = base + (long long)r * stride + hi;        // K = hi..hi+7
  const _Float16* p1 = p0 + 16;                                  // K = 16+hi..16+hi+7
  v8h a0 = *(const v8h*)p0;
  v8h a1 = *(const v8h*)p1;
  v16h f;
#pragma unroll
  for (int j = 0; j < 8; ++j) { f[j] = a0[j]; f[j + 8] = a1[j]; }
  return f;
}

// B-matrix 32x16 f16: lanes 0-15 hold column N=lane with K = {0..7, 16..23};
// lanes 16-31 hold column N=lane-16 with K = {8..15, 24..31}.
// Source stored "column-major" as [16 x K] rows (row = N) with stride `stride`.
__device__ __forceinline__ v16h load_fragB(const _Float16* __restrict__ base, int stride) {
  int lane = threadIdx.x & 31;
  int n    = lane & 15;
  int kb   = (lane >> 4) << 3;            // 0 or 8
  const _Float16* p = base + (long long)n * stride + kb;
  v8h b0 = *(const v8h*)p;
  v8h b1 = *(const v8h*)(p + 16);
  v16h f;
#pragma unroll
  for (int j = 0; j < 8; ++j) { f[j] = b0[j]; f[j + 8] = b1[j]; }
  return f;
}

__device__ __forceinline__ float gelu_exact(float z) {
  return 0.5f * z * (1.0f + erff(z * 0.70710678118654752f));
}

// ---------------------------------------------------------------------------
// 1) depthwise 3x3 conv + BatchNorm (eval) -> f16, position-major [b][p][c]
// ---------------------------------------------------------------------------
__global__ void dwconv_bn_kernel(const float* __restrict__ x,      // [B,C,64,64]
                                 const float* __restrict__ dw,     // [C,1,3,3]
                                 const float* __restrict__ g,
                                 const float* __restrict__ beta,
                                 const float* __restrict__ mean,
                                 const float* __restrict__ var,
                                 _Float16* __restrict__ out,       // [B, OP, C]
                                 int stride_, int OW) {
  int OP = OW * OW;
  long long idx   = (long long)blockIdx.x * blockDim.x + threadIdx.x;
  long long total = (long long)NB * OP * NC;
  if (idx >= total) return;
  int c = (int)(idx % NC);
  int p = (int)((idx / NC) % OP);
  int b = (int)(idx / ((long long)NC * OP));
  int oy = p / OW, ox = p % OW;
  const float* xp = x + ((long long)b * NC + c) * (NHW * NHW);
  float acc = 0.0f;
#pragma unroll
  for (int dy = 0; dy < 3; ++dy) {
    int yy = oy * stride_ + dy - 1;
    if (yy < 0 || yy >= NHW) continue;
#pragma unroll
    for (int dx = 0; dx < 3; ++dx) {
      int xx = ox * stride_ + dx - 1;
      if (xx < 0 || xx >= NHW) continue;
      acc += xp[yy * NHW + xx] * dw[c * 9 + dy * 3 + dx];
    }
  }
  float sc = g[c] * rsqrtf(var[c] + 1e-5f);
  out[idx] = (_Float16)((acc - mean[c]) * sc + beta[c]);
}

// ---------------------------------------------------------------------------
// f32 -> f16 copy (weights)
// ---------------------------------------------------------------------------
__global__ void f32_to_f16_kernel(const float* __restrict__ in, _Float16* __restrict__ out, int n) {
  int i = blockIdx.x * blockDim.x + threadIdx.x;
  if (i < n) out[i] = (_Float16)in[i];
}

// ---------------------------------------------------------------------------
// 2) pointwise 1x1 conv as GEMM: Z[o,p] = GELU(sum_c W[o,c]*Y[p,c] + bias[o])
//    Register-blocked: one wave computes a 32(o) x 64(p) tile.
//    Per K-step: 2 A-fragments + 4 B-fragments feed 8 WMMAs.
//    mode 0: f32 out, position-major [b][p][o]  (q, feeds softmax over d)
//    mode 1: f32 out, channel-major  [b][o][p]  (k, feeds softmax over p)
//    mode 2: f16 out, channel-major  [b][o][p]  (v, feeds ctx GEMM B-matrix)
// ---------------------------------------------------------------------------
__global__ void pw_gemm_gelu_kernel(const _Float16* __restrict__ W,   // [C][C] f16 (row o)
                                    const _Float16* __restrict__ Y,   // [B][P][C] f16
                                    const float* __restrict__ bias,   // [C]
                                    float* __restrict__ outF,
                                    _Float16* __restrict__ outH,
                                    int P, int mode) {
  int tilesP    = P >> 6;                 // 64-wide position blocks
  int tilesO    = NC >> 5;                // 32-wide channel blocks
  int tilesPerB = tilesO * tilesP;
  int tile = blockIdx.x * blockDim.y + threadIdx.y;
  if (tile >= NB * tilesPerB) return;
  int b  = tile / tilesPerB;
  int r  = tile % tilesPerB;
  int o0 = (r / tilesP) << 5;
  int p0 = (r % tilesP) << 6;

  const _Float16* Wb = W + (long long)o0 * NC;
  const _Float16* Yb = Y + ((long long)b * P + p0) * NC;
  v8f acc[2][4] = {};
  for (int k0 = 0; k0 < NC; k0 += 32) {
    v16h a0 = load_fragA(Wb + k0, NC);
    v16h a1 = load_fragA(Wb + 16LL * NC + k0, NC);
    v16h b0 = load_fragB(Yb + k0, NC);
    v16h b1 = load_fragB(Yb + 16LL * NC + k0, NC);
    v16h b2 = load_fragB(Yb + 32LL * NC + k0, NC);
    v16h b3 = load_fragB(Yb + 48LL * NC + k0, NC);
    acc[0][0] = WMMA_F16(a0, b0, acc[0][0]);
    acc[0][1] = WMMA_F16(a0, b1, acc[0][1]);
    acc[0][2] = WMMA_F16(a0, b2, acc[0][2]);
    acc[0][3] = WMMA_F16(a0, b3, acc[0][3]);
    acc[1][0] = WMMA_F16(a1, b0, acc[1][0]);
    acc[1][1] = WMMA_F16(a1, b1, acc[1][1]);
    acc[1][2] = WMMA_F16(a1, b2, acc[1][2]);
    acc[1][3] = WMMA_F16(a1, b3, acc[1][3]);
  }
  int lane = threadIdx.x & 31;
  int col  = lane & 15;                // N -> position
  int rb   = (lane >> 4) << 3;         // M base -> output channel
#pragma unroll
  for (int io = 0; io < 2; ++io) {
#pragma unroll
    for (int jp = 0; jp < 4; ++jp) {
      int p = p0 + (jp << 4) + col;
#pragma unroll
      for (int i = 0; i < 8; ++i) {
        int o = o0 + (io << 4) + rb + i;
        float z = gelu_exact(acc[io][jp][i] + bias[o]);
        if (mode == 0)      outF[((long long)b * P + p) * NC + o] = z;
        else if (mode == 1) outF[((long long)b * NC + o) * P + p] = z;
        else                outH[((long long)b * NC + o) * P + p] = (_Float16)z;
      }
    }
  }
}

// ---------------------------------------------------------------------------
// 3) softmax over head-dim (q): groups of 64 contiguous f32 -> f16
// ---------------------------------------------------------------------------
__global__ void softmax_d_kernel(const float* __restrict__ in,     // [B][PQ][C]
                                 _Float16* __restrict__ out) {     // [B][PQ][C]
  int idx = blockIdx.x * blockDim.x + threadIdx.x;                 // (b,p,h)
  if (idx >= NB * PQ * NH) return;
  const float* p = in + (long long)idx * 64;
  float mx = -1e30f;
  for (int i = 0; i < 64; ++i) mx = fmaxf(mx, p[i]);
  float s = 0.0f;
  for (int i = 0; i < 64; ++i) s += expf(p[i] - mx);
  float inv = 1.0f / s;
  _Float16* o = out + (long long)idx * 64;
  for (int i = 0; i < 64; ++i) o[i] = (_Float16)(expf(p[i] - mx) * inv);
}

// ---------------------------------------------------------------------------
// 4) softmax over sequence (k): one wave per (b,c) row of 1024 f32 -> f16
// ---------------------------------------------------------------------------
__global__ void softmax_p_kernel(const float* __restrict__ in,     // [B][C][PK]
                                 _Float16* __restrict__ out) {     // [B][C][PK]
  int row  = blockIdx.x;            // b*C + c
  int lane = threadIdx.x;           // 0..31
  const float* p = in + (long long)row * PK;
  float mx = -1e30f;
  for (int i = lane; i < PK; i += 32) mx = fmaxf(mx, p[i]);
#pragma unroll
  for (int d = 16; d > 0; d >>= 1) mx = fmaxf(mx, __shfl_xor(mx, d, 32));
  float s = 0.0f;
  for (int i = lane; i < PK; i += 32) s += expf(p[i] - mx);
#pragma unroll
  for (int d = 16; d > 0; d >>= 1) s += __shfl_xor(s, d, 32);
  float inv = 1.0f / s;
  _Float16* o = out + (long long)row * PK;
  for (int i = lane; i < PK; i += 32) o[i] = (_Float16)(expf(p[i] - mx) * inv);
}

// ---------------------------------------------------------------------------
// 5) ctx GEMM per (b,h): ctx[d,e] = sum_l kh[d,l] * v[e,l]   (K = 1024)
//    Register-blocked 32x32 per wave: 2 A + 2 B fragments -> 4 WMMAs / K-step.
//    kh: [B][C][PK] f16 (rows d), v: [B][C][PK] f16 (rows e, B-matrix)
//    Writes ctx transposed: ctxT[b][h][e][d] f16 (B-matrix for final GEMM).
// ---------------------------------------------------------------------------
__global__ void ctx_gemm_kernel(const _Float16* __restrict__ kh,
                                const _Float16* __restrict__ vch,
                                _Float16* __restrict__ ctxT) {
  int tile = blockIdx.x * blockDim.y + threadIdx.y;   // 0..255
  if (tile >= NB * NH * 4) return;
  int t  = tile & 3;
  int bh = tile >> 2;                 // b*8 + h
  int b  = bh >> 3, h = bh & 7;
  int d0 = (t >> 1) << 5;             // 0 or 32
  int e0 = (t & 1) << 5;              // 0 or 32
  const _Float16* A  = kh  + ((long long)b * NC + h * ND + d0) * PK;
  const _Float16* Bm = vch + ((long long)b * NC + h * ND + e0) * PK;
  v8f acc[2][2] = {};
  for (int k0 = 0; k0 < PK; k0 += 32) {
    v16h a0 = load_fragA(A + k0, PK);
    v16h a1 = load_fragA(A + 16LL * PK + k0, PK);
    v16h b0 = load_fragB(Bm + k0, PK);
    v16h b1 = load_fragB(Bm + 16LL * PK + k0, PK);
    acc[0][0] = WMMA_F16(a0, b0, acc[0][0]);
    acc[0][1] = WMMA_F16(a0, b1, acc[0][1]);
    acc[1][0] = WMMA_F16(a1, b0, acc[1][0]);
    acc[1][1] = WMMA_F16(a1, b1, acc[1][1]);
  }
  int lane = threadIdx.x & 31;
  int col  = lane & 15;               // N -> e
  int rb   = (lane >> 4) << 3;        // M -> d
#pragma unroll
  for (int id = 0; id < 2; ++id) {
#pragma unroll
    for (int ie = 0; ie < 2; ++ie) {
      _Float16* dst = ctxT + ((long long)bh * ND + e0 + (ie << 4) + col) * ND
                           + d0 + (id << 4) + rb;
#pragma unroll
      for (int i = 0; i < 8; ++i) dst[i] = (_Float16)acc[id][ie][i];
    }
  }
}

// ---------------------------------------------------------------------------
// 6) out GEMM per (b,h): att[p,e] = sum_d qh[p,d] * ctx[d,e]   (K = 64)
//    Register-blocked 64(p) x 16(e): one shared B fragment per K-step
//    feeds 4 WMMAs. Fused epilogue: y = ww0*att + ww1*x ; BN1d -> d_out.
// ---------------------------------------------------------------------------
__global__ void out_gemm_fused_kernel(const _Float16* __restrict__ qh,    // [B][PQ][C]
                                      const _Float16* __restrict__ ctxT,  // [B][H][ND][ND]
                                      const float* __restrict__ x,        // [B][C][L]
                                      const float* __restrict__ add_w,    // [2]
                                      const float* __restrict__ g1,
                                      const float* __restrict__ b1,
                                      const float* __restrict__ m1,
                                      const float* __restrict__ v1,
                                      float* __restrict__ out) {
  int tile = blockIdx.x * blockDim.y + threadIdx.y;   // 0..2047
  if (tile >= NB * NH * (PQ / 64) * (ND / 16)) return;
  int et = tile & 3;
  int pt = (tile >> 2) & 63;
  int bh = tile >> 8;
  int b  = bh >> 3, h = bh & 7;
  int p0 = pt << 6, e0 = et << 4;

  const _Float16* A  = qh + ((long long)b * PQ + p0) * NC + h * ND;
  const _Float16* Bm = ctxT + ((long long)bh * ND + e0) * ND;
  v8f acc[4] = {};
#pragma unroll
  for (int k0 = 0; k0 < ND; k0 += 32) {
    v16h bf = load_fragB(Bm + k0, ND);
#pragma unroll
    for (int j = 0; j < 4; ++j) {
      v16h a = load_fragA(A + (long long)(j << 4) * NC + k0, NC);
      acc[j] = WMMA_F16(a, bf, acc[j]);
    }
  }
  float w0 = fmaxf(add_w[0], 0.0f), w1 = fmaxf(add_w[1], 0.0f);
  float inv = 1.0f / (w0 + w1 + 1e-12f);
  w0 *= inv; w1 *= inv;

  int lane = threadIdx.x & 31;
  int col  = lane & 15;               // N -> e
  int rb   = (lane >> 4) << 3;        // M -> position
  int c    = h * ND + e0 + col;
  float sc = g1[c] * rsqrtf(v1[c] + 1e-5f);
  float mm = m1[c], bb = b1[c];
#pragma unroll
  for (int j = 0; j < 4; ++j) {
    long long base = ((long long)b * NC + c) * NL + p0 + (j << 4) + rb;
#pragma unroll
    for (int i = 0; i < 8; ++i) {
      float y = w0 * acc[j][i] + w1 * x[base + i];
      out[base + i] = (y - mm) * sc + bb;
    }
  }
}

// ---------------------------------------------------------------------------
// Host-side launcher
// ---------------------------------------------------------------------------
extern "C" void kernel_launch(void* const* d_in, const int* in_sizes, int n_in,
                              void* d_out, int out_size, void* d_ws, size_t ws_size,
                              hipStream_t stream) {
  (void)in_sizes; (void)n_in; (void)out_size; (void)ws_size;
  const float* x      = (const float*)d_in[0];
  const float* q_dw   = (const float*)d_in[1];
  const float* q_bn_g = (const float*)d_in[2];
  const float* q_bn_b = (const float*)d_in[3];
  const float* q_bn_m = (const float*)d_in[4];
  const float* q_bn_v = (const float*)d_in[5];
  const float* q_pw_w = (const float*)d_in[6];
  const float* q_pw_b = (const float*)d_in[7];
  const float* k_dw   = (const float*)d_in[8];
  const float* k_bn_g = (const float*)d_in[9];
  const float* k_bn_b = (const float*)d_in[10];
  const float* k_bn_m = (const float*)d_in[11];
  const float* k_bn_v = (const float*)d_in[12];
  const float* k_pw_w = (const float*)d_in[13];
  const float* k_pw_b = (const float*)d_in[14];
  const float* v_dw   = (const float*)d_in[15];
  const float* v_bn_g = (const float*)d_in[16];
  const float* v_bn_b = (const float*)d_in[17];
  const float* v_bn_m = (const float*)d_in[18];
  const float* v_bn_v = (const float*)d_in[19];
  const float* v_pw_w = (const float*)d_in[20];
  const float* v_pw_b = (const float*)d_in[21];
  const float* add_w  = (const float*)d_in[22];
  const float* bn1_g  = (const float*)d_in[23];
  const float* bn1_b  = (const float*)d_in[24];
  const float* bn1_m  = (const float*)d_in[25];
  const float* bn1_v  = (const float*)d_in[26];
  float* out = (float*)d_out;

  // Workspace layout (bytes), regions aliased along the dependency chain:
  //  A [0,        33554432): Yq f16 -> qh f16
  //  B [33554432, 100663296): Zq f32; later vch f16 (@+0) and ctxT f16 (@+16MiB)
  //  C [100663296,109051904): Yk f16 -> kh f16
  //  D [109051904,125829120): Zk f32; later Yv f16
  //  E [125829120,127401984): Wq/Wk/Wv f16
  char* ws = (char*)d_ws;
  _Float16* Yq   = (_Float16*)(ws);
  _Float16* qh   = (_Float16*)(ws);
  float*    ZqF  = (float*)   (ws + 33554432ULL);
  _Float16* vch  = (_Float16*)(ws + 33554432ULL);
  _Float16* ctxT = (_Float16*)(ws + 33554432ULL + 16777216ULL);
  _Float16* Yk   = (_Float16*)(ws + 100663296ULL);
  _Float16* kh   = (_Float16*)(ws + 100663296ULL);
  float*    ZkF  = (float*)   (ws + 109051904ULL);
  _Float16* Yv   = (_Float16*)(ws + 109051904ULL);
  _Float16* Wq   = (_Float16*)(ws + 125829120ULL);
  _Float16* Wk   = Wq + 262144;
  _Float16* Wv   = Wk + 262144;

  // Weight conversion (f32 -> f16), 512*512 each
  f32_to_f16_kernel<<<1024, 256, 0, stream>>>(q_pw_w, Wq, NC * NC);
  f32_to_f16_kernel<<<1024, 256, 0, stream>>>(k_pw_w, Wk, NC * NC);
  f32_to_f16_kernel<<<1024, 256, 0, stream>>>(v_pw_w, Wv, NC * NC);

  dim3 wblk(32, 4, 1);

  // ---- q branch ----
  dwconv_bn_kernel<<<65536, 256, 0, stream>>>(x, q_dw, q_bn_g, q_bn_b, q_bn_m, q_bn_v,
                                              Yq, 1, 64);
  // tiles: 8 * (512/32) * (4096/64) = 8192 -> 2048 blocks of 4 waves
  pw_gemm_gelu_kernel<<<2048, wblk, 0, stream>>>(Wq, Yq, q_pw_b, ZqF, nullptr, PQ, 0);
  softmax_d_kernel<<<1024, 256, 0, stream>>>(ZqF, qh);

  // ---- k branch ----
  dwconv_bn_kernel<<<16384, 256, 0, stream>>>(x, k_dw, k_bn_g, k_bn_b, k_bn_m, k_bn_v,
                                              Yk, 2, 32);
  // tiles: 8 * 16 * (1024/64) = 2048 -> 512 blocks
  pw_gemm_gelu_kernel<<<512, wblk, 0, stream>>>(Wk, Yk, k_pw_b, ZkF, nullptr, PK, 1);
  softmax_p_kernel<<<NB * NC, 32, 0, stream>>>(ZkF, kh);

  // ---- v branch ----
  dwconv_bn_kernel<<<16384, 256, 0, stream>>>(x, v_dw, v_bn_g, v_bn_b, v_bn_m, v_bn_v,
                                              Yv, 2, 32);
  pw_gemm_gelu_kernel<<<512, wblk, 0, stream>>>(Wv, Yv, v_pw_b, nullptr, vch, PK, 2);

  // ---- attention GEMMs ----
  // ctx tiles: 8*8*4 = 256 -> 64 blocks
  ctx_gemm_kernel<<<64, wblk, 0, stream>>>(kh, vch, ctxT);
  // out tiles: 8*8*64*4 = 16384; 512 blocks of 4 waves run tiles 0..2047
  out_gemm_fused_kernel<<<512, wblk, 0, stream>>>(qh, ctxT, x, add_w,
                                                  bn1_g, bn1_b, bn1_m, bn1_v, out);
}
